// LaneNet_38319698215133
// MI455X (gfx1250) — matrix operand order repaired and checked by reading, not compiled, not run on hardware
//
#include <hip/hip_runtime.h>

#ifndef __has_builtin
#define __has_builtin(x) 0
#endif

typedef _Float16 h4  __attribute__((ext_vector_type(4)));
typedef _Float16 h8  __attribute__((ext_vector_type(8)));
typedef _Float16 v16h __attribute__((ext_vector_type(16)));
typedef float    v8f  __attribute__((ext_vector_type(8)));
typedef int      v4i_ __attribute__((ext_vector_type(4)));

#define D   128
#define NB  4
#define NS  6
#define MT  8            // M-tiles (of 16 rows) per workgroup
#define GN_EPS 1e-5f

#define AS1 __attribute__((address_space(1)))
#define AS3 __attribute__((address_space(3)))

// Async global -> LDS copy selection (CDNA5 ASYNCcnt path)
#if __has_builtin(__builtin_amdgcn_global_load_async_to_lds_b128)
#define ASYNC_MODE 1
#elif __has_builtin(__builtin_amdgcn_load_to_lds)
#define ASYNC_MODE 2
#else
#define ASYNC_MODE 0
#endif

// ---------------- helpers ----------------

// A fragment (16x32 f16, ISA layout): per lane halves [k0..k0+7] and [k0+16..k0+23]
__device__ __forceinline__ v16h load_fragA(const _Float16* p) {
  h8 lo = *(const h8*)(p);
  h8 hi = *(const h8*)(p + 16);
  return __builtin_shufflevector(lo, hi, 0,1,2,3,4,5,6,7,8,9,10,11,12,13,14,15);
}

// B fragment (32x16 f16, ISA layout): lane = column, halves [k0..k0+7] and [k0+16..k0+23]
__device__ __forceinline__ v16h load_fragB(const _Float16* p) {
  h8 lo = *(const h8*)(p);
  h8 hi = *(const h8*)(p + 16);
  return __builtin_shufflevector(lo, hi, 0,1,2,3,4,5,6,7,8,9,10,11,12,13,14,15);
}

// Native fp32 global atomic add (no-return form -> STOREcnt, fire & forget)
__device__ __forceinline__ void atomic_add_f32(float* p, float v) {
  asm volatile("global_atomic_add_f32 %0, %1, off" :: "v"(p), "v"(v) : "memory");
}

// Async copy of 16 bytes global -> LDS (per lane)
__device__ __forceinline__ void async_copy_b128(const _Float16* gp, _Float16* lp) {
#if ASYNC_MODE == 1
  __builtin_amdgcn_global_load_async_to_lds_b128((v4i_*)gp, (AS3 v4i_*)lp, 0, 0);
#elif ASYNC_MODE == 2
  __builtin_amdgcn_load_to_lds((v4i_*)gp, (AS3 v4i_*)lp, 16, 0, 0);
#else
  *(h8*)lp = *(const h8*)gp;
#endif
}

__device__ __forceinline__ void wait_async0() {
#if ASYNC_MODE
#if __has_builtin(__builtin_amdgcn_s_wait_asynccnt)
  __builtin_amdgcn_s_wait_asynccnt(0);
#else
  asm volatile("s_wait_asynccnt 0" ::: "memory");
#endif
#endif
}

__device__ __forceinline__ float wave_sum32(float v) {
  #pragma unroll
  for (int m = 16; m > 0; m >>= 1) v += __shfl_xor(v, m, 32);
  return v;
}

#define WMMA_F16(a, b, c) \
  __builtin_amdgcn_wmma_f32_16x16x32_f16(false, (a), false, (b), (short)0, (c), false, false)

// ---------------- weight conversion ----------------

__global__ __launch_bounds__(256) void f32_to_f16_kernel(const float* __restrict__ src,
                                                         _Float16* __restrict__ dst, int n) {
  int i = blockIdx.x * 256 + threadIdx.x;
  if (i < n) dst[i] = (_Float16)src[i];
}

// ---------------- input branch stage 1: 2->128 MLPs ----------------

__global__ __launch_bounds__(256) void input_stage1_kernel(
    const float* __restrict__ coords,
    const float* __restrict__ in_w1, const float* __restrict__ in_b1,
    const float* __restrict__ seg_w1, const float* __restrict__ seg_b1,
    _Float16* __restrict__ hin, _Float16* __restrict__ hseg, int N) {
  int gid = blockIdx.x * 256 + threadIdx.x;
  if (gid >= N * 16) return;
  int row = gid >> 4;
  int cg  = gid & 15;
  float4 c4 = *(const float4*)(coords + (size_t)row * 4);
  float cx = 0.5f * (c4.x + c4.z), cy = 0.5f * (c4.y + c4.w);
  float dx = c4.z - c4.x,          dy = c4.w - c4.y;
  h8 a, b;
  #pragma unroll
  for (int j = 0; j < 8; ++j) {
    int o = cg * 8 + j;
    float hi = fmaxf(in_w1[o * 2] * cx + in_w1[o * 2 + 1] * cy + in_b1[o], 0.f);
    float hs = fmaxf(seg_w1[o * 2] * dx + seg_w1[o * 2 + 1] * dy + seg_b1[o], 0.f);
    a[j] = (_Float16)hi;
    b[j] = (_Float16)hs;
  }
  *(h8*)(hin  + (size_t)row * D + cg * 8) = a;
  *(h8*)(hseg + (size_t)row * D + cg * 8) = b;
}

// ---------------- dense GEMM: out[N,128] = A16[N,128] @ W^T (W is [out][in]) ----------------
// Register double-buffered software pipeline: prefetch A tile mt+1 while computing mt.

__global__ __launch_bounds__(256) void dense_gemm_kernel(
    const _Float16* __restrict__ A16, const _Float16* __restrict__ W,
    float* __restrict__ out, int mtiles) {
  const int lane = threadIdx.x & 31;
  const int wave = threadIdx.x >> 5;          // N-tile 0..7
  const int col  = lane & 15;
  const int n    = wave * 16 + col;
  const int kb16 = (lane >> 4) * 8;           // B K-chunk
  const int koff = (lane >> 4) * 8;           // A K-chunk
  const int arow = lane & 15;
  const int mhalf = (lane >> 4) * 8;

  const _Float16* wr = W + (size_t)n * D;
  v16h b0 = load_fragB(wr + 0 * 32 + kb16);
  v16h b1 = load_fragB(wr + 1 * 32 + kb16);
  v16h b2 = load_fragB(wr + 2 * 32 + kb16);
  v16h b3 = load_fragB(wr + 3 * 32 + kb16);

  int mt0 = blockIdx.x * MT;
  int mt1 = min(mt0 + MT, mtiles);

#define LOAD_A(d0, d1, d2, d3, mt)                                            \
  { const _Float16* ap = A16 + (size_t)((mt) * 16 + arow) * D + koff;         \
    d0 = load_fragA(ap + 0 * 32); d1 = load_fragA(ap + 1 * 32);               \
    d2 = load_fragA(ap + 2 * 32); d3 = load_fragA(ap + 3 * 32); }

#define COMP_STORE(a0, a1, a2, a3, mt)                                        \
  { v8f c = {};                                                               \
    c = WMMA_F16(a0, b0, c); c = WMMA_F16(a1, b1, c);                         \
    c = WMMA_F16(a2, b2, c); c = WMMA_F16(a3, b3, c);                         \
    float* op = out + (size_t)((mt) * 16 + mhalf) * D + n;                    \
    _Pragma("unroll")                                                         \
    for (int r = 0; r < 8; ++r) op[(size_t)r * D] = c[r]; }

  v16h x0, x1, x2, x3, y0, y1, y2, y3;
  int mt = mt0;
  LOAD_A(x0, x1, x2, x3, mt);
  while (mt < mt1) {
    if (mt + 1 < mt1) LOAD_A(y0, y1, y2, y3, mt + 1);
    COMP_STORE(x0, x1, x2, x3, mt);
    ++mt;
    if (mt >= mt1) break;
    if (mt + 1 < mt1) LOAD_A(x0, x1, x2, x3, mt + 1);
    COMP_STORE(y0, y1, y2, y3, mt);
    ++mt;
  }
#undef LOAD_A
#undef COMP_STORE
}

// ---------------- edge gather-GEMM-scatter ----------------
// grid.y in [0,12): s = y>>1 (scale), dir = y&1 (0=pre: src=c1 dst=c0, 1=suc: src=c0 dst=c1)
// Double-buffered LDS staging; gather uses async global->LDS when available.

__global__ __launch_bounds__(256) void edge_gemm_kernel(
    const _Float16* __restrict__ feats16,
    const _Float16* __restrict__ preW,   // [NS][128][128] f16 (this block)
    const _Float16* __restrict__ sucW,   // [NS][128][128] f16 (this block)
    const long long* __restrict__ conns, // [NS+1][E][2] int64
    float* __restrict__ temp, int E) {
  const int s   = blockIdx.y >> 1;
  const int dir = blockIdx.y & 1;
  const _Float16* W = (dir ? sucW : preW) + (size_t)s * D * D;

  const int lane = threadIdx.x & 31;
  const int wave = threadIdx.x >> 5;
  const int col  = lane & 15;
  const int n    = wave * 16 + col;
  const int kb16 = (lane >> 4) * 8;
  const int koff = (lane >> 4) * 8;
  const int arow = lane & 15;
  const int mhalf = (lane >> 4) * 8;

  const _Float16* wr = W + (size_t)n * D;
  v16h b0 = load_fragB(wr + 0 * 32 + kb16);
  v16h b1 = load_fragB(wr + 1 * 32 + kb16);
  v16h b2 = load_fragB(wr + 2 * 32 + kb16);
  v16h b3 = load_fragB(wr + 3 * 32 + kb16);

  __shared__ _Float16 As[2][16 * D];   // double-buffered gathered A tiles
  __shared__ int dstS[2][16];

  const int mtiles = E >> 4;
  int mt0 = blockIdx.x * MT;
  int mt1 = min(mt0 + MT, mtiles);

  const int ldm = threadIdx.x >> 4;   // edge row 0..15
  const int ldc = threadIdx.x & 15;   // 8-half chunk 0..15
  const long long* cbase = conns + (size_t)(s + 1) * E * 2;

#define STAGE(mt, buf)                                                        \
  { int e = (mt) * 16 + ldm;                                                  \
    const long long* ce = cbase + (size_t)e * 2;                              \
    long long src = dir ? ce[0] : ce[1];                                      \
    if (ldc == 0) dstS[buf][ldm] = (int)(dir ? ce[1] : ce[0]);                \
    async_copy_b128(&feats16[(size_t)src * D + ldc * 8],                      \
                    &As[buf][ldm * D + ldc * 8]); }

  int p = 0;
  STAGE(mt0, 0);
  for (int mt = mt0; mt < mt1; ++mt) {
    wait_async0();        // our async LDS writes (this wave) complete
    __syncthreads();      // tile p staged by all waves; prev reads of p^1 done
    if (mt + 1 < mt1) STAGE(mt + 1, p ^ 1);

    const _Float16* ap = &As[p][arow * D + koff];
    v16h a0 = load_fragA(ap + 0 * 32);
    v16h a1 = load_fragA(ap + 1 * 32);
    v16h a2 = load_fragA(ap + 2 * 32);
    v16h a3 = load_fragA(ap + 3 * 32);
    v8f c = {};
    c = WMMA_F16(a0, b0, c);
    c = WMMA_F16(a1, b1, c);
    c = WMMA_F16(a2, b2, c);
    c = WMMA_F16(a3, b3, c);

    #pragma unroll
    for (int r = 0; r < 8; ++r) {
      int drow = dstS[p][mhalf + r];
      atomic_add_f32(&temp[(size_t)drow * D + n], c[r]);
    }
    p ^= 1;
  }
#undef STAGE
}

// ---------------- per-row GroupNorm kernels (one wave32 per row) ----------------

__global__ __launch_bounds__(256) void input_stage2_kernel(
    const float* __restrict__ tA, const float* __restrict__ tB,
    const float* __restrict__ gsA, const float* __restrict__ gbA,
    const float* __restrict__ gsB, const float* __restrict__ gbB,
    float* __restrict__ fOut, _Float16* __restrict__ f16Out, int nRows) {
  const int lane = threadIdx.x & 31;
  const int row  = blockIdx.x * 8 + (threadIdx.x >> 5);
  if (row >= nRows) return;
  const size_t base = (size_t)row * D + lane * 4;
  float4 a4 = *(const float4*)(tA + base);
  float4 b4 = *(const float4*)(tB + base);
  float a[4] = {a4.x, a4.y, a4.z, a4.w};
  float b[4] = {b4.x, b4.y, b4.z, b4.w};
  float sa = 0, qa = 0, sb = 0, qb = 0;
  #pragma unroll
  for (int j = 0; j < 4; ++j) { sa += a[j]; qa += a[j]*a[j]; sb += b[j]; qb += b[j]*b[j]; }
  sa = wave_sum32(sa); qa = wave_sum32(qa);
  sb = wave_sum32(sb); qb = wave_sum32(qb);
  float muA = sa * (1.f / D), muB = sb * (1.f / D);
  float rA = rsqrtf(qa * (1.f / D) - muA * muA + GN_EPS);
  float rB = rsqrtf(qb * (1.f / D) - muB * muB + GN_EPS);
  float4 o4; h4 o16;
  float o[4];
  #pragma unroll
  for (int j = 0; j < 4; ++j) {
    int ch = lane * 4 + j;
    float ga = (a[j] - muA) * rA * gsA[ch] + gbA[ch];
    float gb_ = (b[j] - muB) * rB * gsB[ch] + gbB[ch];
    o[j] = fmaxf(ga + gb_, 0.f);
    o16[j] = (_Float16)o[j];
  }
  o4.x = o[0]; o4.y = o[1]; o4.z = o[2]; o4.w = o[3];
  *(float4*)(fOut + base) = o4;
  *(h4*)(f16Out + base) = o16;
}

__global__ __launch_bounds__(256) void gn_relu_f16_kernel(
    const float* __restrict__ in, const float* __restrict__ gs, const float* __restrict__ gb,
    _Float16* __restrict__ out16, int nRows) {
  const int lane = threadIdx.x & 31;
  const int row  = blockIdx.x * 8 + (threadIdx.x >> 5);
  if (row >= nRows) return;
  const size_t base = (size_t)row * D + lane * 4;
  float4 x4 = *(const float4*)(in + base);
  float x[4] = {x4.x, x4.y, x4.z, x4.w};
  float s = 0, q = 0;
  #pragma unroll
  for (int j = 0; j < 4; ++j) { s += x[j]; q += x[j]*x[j]; }
  s = wave_sum32(s); q = wave_sum32(q);
  float mu = s * (1.f / D);
  float rv = rsqrtf(q * (1.f / D) - mu * mu + GN_EPS);
  h4 o16;
  #pragma unroll
  for (int j = 0; j < 4; ++j) {
    int ch = lane * 4 + j;
    float g = (x[j] - mu) * rv * gs[ch] + gb[ch];
    o16[j] = (_Float16)fmaxf(g, 0.f);
  }
  *(h4*)(out16 + base) = o16;
}

__global__ __launch_bounds__(256) void combine_kernel(
    const float* __restrict__ gin, const float* __restrict__ gs, const float* __restrict__ gb,
    float* __restrict__ resid, _Float16* __restrict__ f16Out,
    float* __restrict__ finalOut, int nRows) {
  const int lane = threadIdx.x & 31;
  const int row  = blockIdx.x * 8 + (threadIdx.x >> 5);
  if (row >= nRows) return;
  const size_t base = (size_t)row * D + lane * 4;
  float4 x4 = *(const float4*)(gin + base);
  float4 r4 = *(const float4*)(resid + base);
  float x[4] = {x4.x, x4.y, x4.z, x4.w};
  float r[4] = {r4.x, r4.y, r4.z, r4.w};
  float s = 0, q = 0;
  #pragma unroll
  for (int j = 0; j < 4; ++j) { s += x[j]; q += x[j]*x[j]; }
  s = wave_sum32(s); q = wave_sum32(q);
  float mu = s * (1.f / D);
  float rv = rsqrtf(q * (1.f / D) - mu * mu + GN_EPS);
  float4 o4; h4 o16;
  float o[4];
  #pragma unroll
  for (int j = 0; j < 4; ++j) {
    int ch = lane * 4 + j;
    float g = (x[j] - mu) * rv * gs[ch] + gb[ch];
    o[j] = fmaxf(g + r[j], 0.f);
    o16[j] = (_Float16)o[j];
  }
  o4.x = o[0]; o4.y = o[1]; o4.z = o[2]; o4.w = o[3];
  *(float4*)(resid + base) = o4;
  *(h4*)(f16Out + base) = o16;
  if (finalOut) *(float4*)(finalOut + base) = o4;
}

// ---------------- host ----------------

extern "C" void kernel_launch(void* const* d_in, const int* in_sizes, int n_in,
                              void* d_out, int out_size, void* d_ws, size_t ws_size,
                              hipStream_t stream) {
  const float*     coords    = (const float*)d_in[0];
  const long long* conns     = (const long long*)d_in[1];
  const float*     in_w1     = (const float*)d_in[2];
  const float*     in_b1     = (const float*)d_in[3];
  const float*     in_w2     = (const float*)d_in[4];
  const float*     in_gn_s   = (const float*)d_in[5];
  const float*     in_gn_b   = (const float*)d_in[6];
  const float*     seg_w1    = (const float*)d_in[7];
  const float*     seg_b1    = (const float*)d_in[8];
  const float*     seg_w2    = (const float*)d_in[9];
  const float*     seg_gn_s  = (const float*)d_in[10];
  const float*     seg_gn_b  = (const float*)d_in[11];
  const float*     center_w  = (const float*)d_in[12];
  const float*     pre_w     = (const float*)d_in[13];
  const float*     suc_w     = (const float*)d_in[14];
  const float*     gn_s      = (const float*)d_in[15];
  const float*     gn_b      = (const float*)d_in[16];
  const float*     lwgn_w    = (const float*)d_in[17];
  const float*     lwgn_gn_s = (const float*)d_in[18];
  const float*     lwgn_gn_b = (const float*)d_in[19];

  const int N = in_sizes[0] / 4;            // coords [N,2,2]
  const int E = in_sizes[1] / ((NS + 1) * 2);
  const int mtN = N / 16;
  const int mtE = E / 16;

  char* ws = (char*)d_ws;
  size_t off = 0;
  auto walloc = [&](size_t bytes) -> void* {
    void* p = ws + off;
    off += (bytes + 255) & ~(size_t)255;
    return p;
  };
  float*    F32A = (float*)walloc((size_t)N * D * 4);    // temp / gemm out
  float*    F32B = (float*)walloc((size_t)N * D * 4);    // feats f32 (= residual)
  _Float16* F16A = (_Float16*)walloc((size_t)N * D * 2); // feats f16
  _Float16* F16B = (_Float16*)walloc((size_t)N * D * 2); // hidden / post-GN f16
  _Float16* W16  = (_Float16*)walloc((size_t)58 * D * D * 2);

  _Float16* in_w2_16  = W16;
  _Float16* seg_w2_16 = W16 + (size_t)1  * D * D;
  _Float16* center16  = W16 + (size_t)2  * D * D;   // 4 mats
  _Float16* pre16     = W16 + (size_t)6  * D * D;   // 24 mats
  _Float16* suc16     = W16 + (size_t)30 * D * D;   // 24 mats
  _Float16* lwgn16    = W16 + (size_t)54 * D * D;   // 4 mats

  auto conv = [&](const float* src, _Float16* dst, int n) {
    f32_to_f16_kernel<<<(n + 255) / 256, 256, 0, stream>>>(src, dst, n);
  };
  conv(in_w2,    in_w2_16,  D * D);
  conv(seg_w2,   seg_w2_16, D * D);
  conv(center_w, center16,  NB * D * D);
  conv(pre_w,    pre16,     NB * NS * D * D);
  conv(suc_w,    suc16,     NB * NS * D * D);
  conv(lwgn_w,   lwgn16,    NB * D * D);

  // input branch
  input_stage1_kernel<<<(N * 16 + 255) / 256, 256, 0, stream>>>(
      coords, in_w1, in_b1, seg_w1, seg_b1, F16A, F16B, N);
  dim3 gD((mtN + MT - 1) / MT);
  dense_gemm_kernel<<<gD, 256, 0, stream>>>(F16A, in_w2_16,  F32A, mtN);
  dense_gemm_kernel<<<gD, 256, 0, stream>>>(F16B, seg_w2_16, F32B, mtN);
  const int gnBlocks = (N + 7) / 8;
  input_stage2_kernel<<<gnBlocks, 256, 0, stream>>>(
      F32A, F32B, in_gn_s, in_gn_b, seg_gn_s, seg_gn_b, F32B, F16A, N);

  dim3 gE((mtE + MT - 1) / MT, 12);
  float* outF = (float*)d_out;
  for (int i = 0; i < NB; ++i) {
    // temp = feats @ center_w[i].T   (stores -> initializes F32A)
    dense_gemm_kernel<<<gD, 256, 0, stream>>>(F16A, center16 + (size_t)i * D * D, F32A, mtN);
    // scatter-add all 6 scales x 2 directions
    edge_gemm_kernel<<<gE, 256, 0, stream>>>(
        F16A, pre16 + (size_t)i * NS * D * D, suc16 + (size_t)i * NS * D * D, conns, F32A, E);
    // a = relu(GN(temp)) -> f16
    gn_relu_f16_kernel<<<gnBlocks, 256, 0, stream>>>(F32A, gn_s + i * D, gn_b + i * D, F16B, N);
    // t2 = a @ lwgn_w[i].T
    dense_gemm_kernel<<<gD, 256, 0, stream>>>(F16B, lwgn16 + (size_t)i * D * D, F32A, mtN);
    // feats = relu(GN(t2) + residual)
    combine_kernel<<<gnBlocks, 256, 0, stream>>>(
        F32A, lwgn_gn_s + i * D, lwgn_gn_b + i * D, F32B, F16A,
        (i == NB - 1) ? outF : nullptr, N);
  }
}
